// MsCorrBlock_37744172597293
// MI455X (gfx1250) — matrix operand order repaired and checked by reading, not compiled, not run on hardware
//
#include <hip/hip_runtime.h>

typedef unsigned short u16;
typedef __attribute__((ext_vector_type(16))) __bf16 v16bf;
typedef __attribute__((ext_vector_type(8)))  float  v8f;

#define NPIX   200704      /* NT*H*W = 64*56*56 */
#define HW     3136
#define WD     56
#define BN_INV 0.9999950000374997f

union FragU { v16bf v; uint4 q[2]; };

static __device__ __forceinline__ u16 f2b(float x) {
    union { float f; unsigned u; } a; a.f = x;
    unsigned u = a.u;
    u = (u + 0x7FFFu + ((u >> 16) & 1u)) >> 16;
    return (u16)u;
}
static __device__ __forceinline__ float b2f(u16 h) {
    union { unsigned u; float f; } a; a.u = ((unsigned)h) << 16;
    return a.f;
}
// A-fragment (16x32 bf16): lane row = lane&15, K = kb..kb+15 contiguous, kb=(lane>>4)*16
static __device__ __forceinline__ v16bf frag_a(const u16* p) {
    FragU f;
    f.q[0] = *(const uint4*)(p);
    f.q[1] = *(const uint4*)(p + 8);
    return f.v;
}
// B-fragment (32x16 bf16): lane col = lane&15, K = ks..ks+15 contiguous, ks=(lane>>4)*16
static __device__ __forceinline__ v16bf frag_b(const u16* p) {
    FragU f;
    f.q[0] = *(const uint4*)(p);
    f.q[1] = *(const uint4*)(p + 8);
    return f.v;
}
static __device__ __forceinline__ v16bf frag_zero() {
    FragU f;
    f.q[0] = make_uint4(0u, 0u, 0u, 0u);
    f.q[1] = make_uint4(0u, 0u, 0u, 0u);
    return f.v;
}
static __device__ __forceinline__ v8f acc_zero() {
    v8f z = {0.f, 0.f, 0.f, 0.f, 0.f, 0.f, 0.f, 0.f};
    return z;
}
static __device__ __forceinline__ v8f wmma_bf16(v16bf a, v16bf b, v8f c) {
    return __builtin_amdgcn_wmma_f32_16x16x32_bf16(false, a, false, b, (short)0, c, false, false);
}

// ---------------------------------------------------------------------------
// Tensor Data Mover: DMA a flat byte range (global -> LDS). Descriptor per
// CDNA5 ISA 08_async_tensor: 2D tensor, 1 row of nbytes/8 8-byte elements.
// Toolchain uses the 6-arg builtin: (g0, g1, g2, g3, g4, cpol).
// ---------------------------------------------------------------------------
#if __has_builtin(__builtin_amdgcn_tensor_load_to_lds)
#define HAVE_TDM 1
typedef unsigned int u32x4 __attribute__((ext_vector_type(4)));
typedef int          i32x8 __attribute__((ext_vector_type(8)));
typedef int          i32x4 __attribute__((ext_vector_type(4)));

static __device__ __forceinline__ void tdm_copy_to_lds(const void* gsrc,
                                                       unsigned lds_byte_off,
                                                       unsigned nbytes) {
    unsigned long long ga = (unsigned long long)gsrc;
    unsigned n8 = nbytes >> 3;                       // 8-byte elements
    u32x4 g0;
    g0[0] = 1u;                                      // count=1 (valid user D#)
    g0[1] = lds_byte_off;                            // lds_addr
    g0[2] = (unsigned)(ga & 0xffffffffu);            // global_addr[31:0]
    g0[3] = (unsigned)((ga >> 32) & 0x01ffffffu) | 0x80000000u; // addr[56:32] | type=2
    i32x8 g1;
    g1[0] = 0x00030000;                              // data_size=3 (8 bytes)
    g1[1] = (int)((n8 & 0xffffu) << 16);             // tensor_dim0[15:0] @ bits[63:48]
    g1[2] = (int)((n8 >> 16) | (1u << 16));          // tensor_dim0[31:16], tensor_dim1=1
    g1[3] = (int)((n8 & 0xffffu) << 16);             // tile_dim0 @ bits[127:112]
    g1[4] = 1;                                       // tile_dim1 = 1
    g1[5] = (int)n8;                                 // tensor_dim0_stride[31:0]
    g1[6] = 0;
    g1[7] = 0;
    i32x4 z4 = {0, 0, 0, 0};
    i32x8 z8 = {0, 0, 0, 0, 0, 0, 0, 0};
    __builtin_amdgcn_tensor_load_to_lds(g0, g1, z4, z4, z8, 0);
}
#endif

// Tell the optimizer that LDS behind `p` was written by the TDM (it cannot
// see the DMA's stores; without this, LDS reads fold to undef).
static __device__ __forceinline__ void lds_written_fence(u16* p) {
    asm volatile("" : "+r"(p) : : "memory");
}

// One-shot weight staging into LDS (wave 0 drives the TDM; barrier publishes).
static __device__ __forceinline__ void stage_weights(const u16* __restrict__ gsrc,
                                                     u16* lds, int nelem) {
#ifdef HAVE_TDM
    if (threadIdx.x < 32) {
        tdm_copy_to_lds(gsrc, 0u, (unsigned)nelem * 2u);
        __builtin_amdgcn_s_wait_tensorcnt(0);
    }
    lds_written_fence(lds);
#else
    for (int i = threadIdx.x; i < nelem; i += 256) lds[i] = gsrc[i];
#endif
    __syncthreads();
}

// ---------------------------------------------------------------------------
// Weight prep: fp32 -> bf16, [N][K] row-major; conv2 re-laid as [tap][o][ci]
// ---------------------------------------------------------------------------
__global__ __launch_bounds__(256) void k_prep(
    const float* __restrict__ wc1f, const float* __restrict__ wc21f,
    const float* __restrict__ wc22f, const float* __restrict__ wc2f,
    const float* __restrict__ wc3f,
    u16* __restrict__ wc1, u16* __restrict__ wc21, u16* __restrict__ wc22,
    u16* __restrict__ wB2, u16* __restrict__ wc3) {
    int i = blockIdx.x * 256 + threadIdx.x;
    if (i < 64 * 256) wc1[i] = f2b(wc1f[i]);
    if (i < 16 * 64)  wc21[i] = f2b(wc21f[i]);
    if (i < 64 * 64) {
        int o = i >> 6, k = i & 63;
        wc22[i] = f2b(k < 49 ? wc22f[o * 49 + k] : 0.f);
    }
    if (i < 9 * 64 * 64) {                 // [tap][o][ci] : 8KB blocks per tap
        int tap = i >> 12;
        int rem = i & 4095;
        int o = rem >> 6, ci = rem & 63;
        wB2[i] = f2b(wc2f[(o * 64 + ci) * 9 + tap]);
    }
    if (i < 256 * 64) wc3[i] = f2b(wc3f[i]);
}

// ---------------------------------------------------------------------------
// Temporal conv (depthwise over T) + layout NCHW f32 -> NHWC bf16 via LDS
// grid: (cchunk 0..7, h 0..55, b*2+wchunk 0..15), 256 threads
// ---------------------------------------------------------------------------
__global__ __launch_bounds__(256) void k_tconv(
    const float* __restrict__ x,
    const float* __restrict__ w1, const float* __restrict__ w3,
    const float* __restrict__ w5, const float* __restrict__ w7,
    u16* __restrict__ tx) {
    __shared__ u16 lds[8][32][29];
    const int c0 = blockIdx.x * 32;
    const int h  = blockIdx.y;
    const int b  = blockIdx.z >> 1;
    const int w0 = (blockIdx.z & 1) * 28;
    const int g  = c0 >> 6;              // block-uniform group
    const int Kg = 2 * g + 1;
    const float* wt = (g == 0) ? w1 : (g == 1) ? w3 : (g == 2) ? w5 : w7;

    const int tid = threadIdx.x;
    const int ww  = tid & 31;
    const int cq  = tid >> 5;

    if (ww < 28) {
        const int wglob = w0 + ww;
#pragma unroll
        for (int cs = 0; cs < 4; ++cs) {
            const int cl = cq * 4 + cs;
            const int c  = c0 + cl;
            const int cg = c & 63;
            float xp[14];
#pragma unroll
            for (int i = 0; i < 3; ++i) { xp[i] = 0.f; xp[11 + i] = 0.f; }
#pragma unroll
            for (int t = 0; t < 8; ++t)
                xp[t + 3] = x[((size_t)(b * 8 + t) * 256 + c) * HW + h * WD + wglob];
            float tap7[7];
#pragma unroll
            for (int d = 0; d < 7; ++d) {
                int j = d - 3 + g;
                tap7[d] = (j >= 0 && j < Kg) ? wt[cg * Kg + j] : 0.f;
            }
#pragma unroll
            for (int t = 0; t < 8; ++t) {
                float s = 0.f;
#pragma unroll
                for (int d = 0; d < 7; ++d) s += tap7[d] * xp[t + d];
                lds[t][cl][ww] = f2b(s);
            }
        }
    }
    __syncthreads();
    const int grp = tid >> 5;
    const int cl  = tid & 31;
    for (int it = 0; it < 28; ++it) {
        int combo = it * 8 + grp;        // 0..223 = 8t * 28w
        int t = combo / 28;
        int w = combo - t * 28;
        tx[(((size_t)(b * 8 + t) * HW) + h * WD + (w0 + w)) * 256 + c0 + cl] = lds[t][cl][w];
    }
}

// ---------------------------------------------------------------------------
// conv1: [NPIX,256] x [256,64] -> res2 bf16  (BN1 + ReLU); B staged via TDM
// ---------------------------------------------------------------------------
__global__ __launch_bounds__(256) void k_conv1(
    const u16* __restrict__ tx, const u16* __restrict__ wb,
    const float* __restrict__ g, const float* __restrict__ bta,
    u16* __restrict__ res2) {
    __shared__ alignas(16) u16 wlds[64 * 256];
    stage_weights(wb, wlds, 64 * 256);
    const int lane = threadIdx.x & 31;
    const int m0 = (blockIdx.x * 8 + (threadIdx.x >> 5)) * 16;
    const int nl = lane & 15;
    const int kbA = (lane >> 4) << 4;
    const int kbB = (lane >> 4) << 4;
    const u16* arow = tx + (size_t)(m0 + nl) * 256 + kbA;
    v8f acc[4];
#pragma unroll
    for (int j = 0; j < 4; ++j) acc[j] = acc_zero();
#pragma unroll
    for (int k0 = 0; k0 < 256; k0 += 32) {
        v16bf a = frag_a(arow + k0);
#pragma unroll
        for (int j = 0; j < 4; ++j)
            acc[j] = wmma_bf16(a, frag_b(&wlds[(j * 16 + nl) * 256 + k0 + kbB]), acc[j]);
    }
    const int mo = (lane >> 4) << 3;
#pragma unroll
    for (int j = 0; j < 4; ++j) {
        int n = j * 16 + nl;
        float s = g[n] * BN_INV, bb = bta[n];
#pragma unroll
        for (int r = 0; r < 8; ++r) {
            float v = fmaxf(acc[j][r] * s + bb, 0.f);
            res2[(size_t)(m0 + mo + r) * 64 + n] = f2b(v);
        }
    }
}

// ---------------------------------------------------------------------------
// conv21: [NPIX,64] x [64,16] -> y fp32  (BN21 + ReLU)
// ---------------------------------------------------------------------------
__global__ __launch_bounds__(256) void k_conv21(
    const u16* __restrict__ res2, const u16* __restrict__ wb,
    const float* __restrict__ g, const float* __restrict__ bta,
    float* __restrict__ y) {
    __shared__ alignas(16) u16 wlds[16 * 64];
    stage_weights(wb, wlds, 16 * 64);
    const int lane = threadIdx.x & 31;
    const int m0 = (blockIdx.x * 8 + (threadIdx.x >> 5)) * 16;
    const int nl = lane & 15;
    const int kbA = (lane >> 4) << 4;
    const int kbB = (lane >> 4) << 4;
    const u16* arow = res2 + (size_t)(m0 + nl) * 64 + kbA;
    v8f acc = acc_zero();
#pragma unroll
    for (int k0 = 0; k0 < 64; k0 += 32) {
        acc = wmma_bf16(frag_a(arow + k0), frag_b(&wlds[nl * 64 + k0 + kbB]), acc);
    }
    const int mo = (lane >> 4) << 3;
    float s = g[nl] * BN_INV, bb = bta[nl];
#pragma unroll
    for (int r = 0; r < 8; ++r) {
        float v = fmaxf(acc[r] * s + bb, 0.f);
        y[(size_t)(m0 + mo + r) * 16 + nl] = v;
    }
}

// ---------------------------------------------------------------------------
// 7x7 correlation over 16 channels + BN22 + ReLU -> corr bf16, K padded to 64
// ---------------------------------------------------------------------------
__global__ __launch_bounds__(256) void k_corr(
    const float* __restrict__ y, const float* __restrict__ g22,
    const float* __restrict__ b22, u16* __restrict__ corr) {
    const int pix = blockIdx.x * 256 + threadIdx.x;
    const int n = pix / HW;
    const int hw = pix - n * HW;
    const int h = hw / WD;
    const int w = hw - h * WD;
    const int b = n >> 3, t = n & 7;
    const int t2 = (t + 1 > 7) ? 7 : t + 1;
    const int n2 = (b << 3) + t2;
    float a[16];
    const float* ap = y + (size_t)pix * 16;
#pragma unroll
    for (int c = 0; c < 16; ++c) a[c] = ap[c];
    const float* ybase = y + (size_t)n2 * HW * 16;
#pragma unroll
    for (int di = 0; di < 7; ++di) {
        const int sh = h + di - 3;
        const bool vh = (unsigned)sh < (unsigned)WD;
#pragma unroll
        for (int dj = 0; dj < 7; ++dj) {
            const int sw = w + dj - 3;
            float dot = 0.f;
            if (vh && (unsigned)sw < (unsigned)WD) {
                const float* bp = ybase + (size_t)(sh * WD + sw) * 16;
#pragma unroll
                for (int c = 0; c < 16; ++c) dot += a[c] * bp[c];
            }
            const int k = di * 7 + dj;
            float v = dot * (1.f / 16.f) * (g22[k] * BN_INV) + b22[k];
            corr[(size_t)pix * 64 + k] = f2b(fmaxf(v, 0.f));
        }
    }
#pragma unroll
    for (int k = 49; k < 64; ++k) corr[(size_t)pix * 64 + k] = 0;
}

// ---------------------------------------------------------------------------
// conv22: [NPIX,64pad] x [64,64] + BN23 + res2 + ReLU -> out2 bf16
// ---------------------------------------------------------------------------
__global__ __launch_bounds__(256) void k_conv22(
    const u16* __restrict__ corr, const u16* __restrict__ wb,
    const float* __restrict__ g, const float* __restrict__ bta,
    const u16* __restrict__ res2, u16* __restrict__ out2) {
    __shared__ alignas(16) u16 wlds[64 * 64];
    stage_weights(wb, wlds, 64 * 64);
    const int lane = threadIdx.x & 31;
    const int m0 = (blockIdx.x * 8 + (threadIdx.x >> 5)) * 16;
    const int nl = lane & 15;
    const int kbA = (lane >> 4) << 4;
    const int kbB = (lane >> 4) << 4;
    const u16* arow = corr + (size_t)(m0 + nl) * 64 + kbA;
    v8f acc[4];
#pragma unroll
    for (int j = 0; j < 4; ++j) acc[j] = acc_zero();
#pragma unroll
    for (int k0 = 0; k0 < 64; k0 += 32) {
        v16bf a = frag_a(arow + k0);
#pragma unroll
        for (int j = 0; j < 4; ++j)
            acc[j] = wmma_bf16(a, frag_b(&wlds[(j * 16 + nl) * 64 + k0 + kbB]), acc[j]);
    }
    const int mo = (lane >> 4) << 3;
#pragma unroll
    for (int j = 0; j < 4; ++j) {
        int n = j * 16 + nl;
        float s = g[n] * BN_INV, bb = bta[n];
#pragma unroll
        for (int r = 0; r < 8; ++r) {
            size_t idx = (size_t)(m0 + mo + r) * 64 + n;
            float v = fmaxf(acc[j][r] * s + bb + b2f(res2[idx]), 0.f);
            out2[idx] = f2b(v);
        }
    }
}

// ---------------------------------------------------------------------------
// conv2: 3x3 SAME as im2col GEMM K=576 + BN2 + ReLU + out2 -> z bf16
// Weights [tap][64o][64ci]; double-buffered TDM: DMA tap t+1 while WMMA on t.
// ---------------------------------------------------------------------------
__global__ __launch_bounds__(256) void k_conv2(
    const u16* __restrict__ res2, const u16* __restrict__ wb /*[9][64][64]*/,
    const float* __restrict__ g, const float* __restrict__ bta,
    const u16* __restrict__ out2, u16* __restrict__ z) {
    __shared__ alignas(16) u16 bl[2][64 * 64];
    const int lane = threadIdx.x & 31;
    const int m0 = (blockIdx.x * 8 + (threadIdx.x >> 5)) * 16;
    const int nl = lane & 15;
    const int kbA = (lane >> 4) << 4;
    const int kbB = (lane >> 4) << 4;
    const int row = m0 + nl;
    const int nim = row / HW;
    const int hw = row - nim * HW;
    const int h = hw / WD;
    const int w = hw - h * WD;
    v8f acc[4];
#pragma unroll
    for (int j = 0; j < 4; ++j) acc[j] = acc_zero();

#ifdef HAVE_TDM
    if (threadIdx.x < 32) tdm_copy_to_lds(wb, 0u, 8192u);     // preload tap 0
#endif
#pragma unroll
    for (int tap = 0; tap < 9; ++tap) {
        const int cur = tap & 1;
#ifdef HAVE_TDM
        if (threadIdx.x < 32) {
            if (tap < 8) {
                tdm_copy_to_lds(wb + (size_t)(tap + 1) * 4096,
                                ((tap + 1) & 1) ? 8192u : 0u, 8192u);
                __builtin_amdgcn_s_wait_tensorcnt(1);   // tap's DMA done, next in flight
            } else {
                __builtin_amdgcn_s_wait_tensorcnt(0);
            }
        }
        lds_written_fence(&bl[cur][0]);
#else
        for (int i = threadIdx.x; i < 4096; i += 256)
            bl[cur][i] = wb[(size_t)tap * 4096 + i];
#endif
        __syncthreads();
        const int sh = h + tap / 3 - 1;
        const int sw = w + tap % 3 - 1;
        const bool ok = ((unsigned)sh < (unsigned)WD) && ((unsigned)sw < (unsigned)WD);
        const u16* abase = res2 + (size_t)(nim * HW + sh * WD + sw) * 64 + kbA;
#pragma unroll
        for (int half = 0; half < 2; ++half) {
            v16bf a;
            if (ok) a = frag_a(abase + half * 32);
            else    a = frag_zero();
#pragma unroll
            for (int j = 0; j < 4; ++j)
                acc[j] = wmma_bf16(a, frag_b(&bl[cur][(j * 16 + nl) * 64 + half * 32 + kbB]), acc[j]);
        }
        __syncthreads();   // all reads of bl[cur] done before its next DMA
    }
    const int mo = (lane >> 4) << 3;
#pragma unroll
    for (int j = 0; j < 4; ++j) {
        int n = j * 16 + nl;
        float s = g[n] * BN_INV, bb = bta[n];
#pragma unroll
        for (int r = 0; r < 8; ++r) {
            size_t idx = (size_t)(m0 + mo + r) * 64 + n;
            float v = fmaxf(acc[j][r] * s + bb, 0.f) + b2f(out2[idx]);
            z[idx] = f2b(v);
        }
    }
}

// ---------------------------------------------------------------------------
// conv3: [NPIX,64] x [64,256] + BN3 + residual + ReLU -> d_out fp32 NCHW
// ---------------------------------------------------------------------------
__global__ __launch_bounds__(256) void k_conv3(
    const u16* __restrict__ zin, const u16* __restrict__ wb /*[256][64]*/,
    const float* __restrict__ g, const float* __restrict__ bta,
    const float* __restrict__ xres, float* __restrict__ out) {
    __shared__ alignas(16) u16 wlds[256 * 64];
    stage_weights(wb, wlds, 256 * 64);
    const int lane = threadIdx.x & 31;
    const int m0 = (blockIdx.x * 8 + (threadIdx.x >> 5)) * 16;
    const int nl = lane & 15;
    const int kbA = (lane >> 4) << 4;
    const int kbB = (lane >> 4) << 4;
    const int nbase = blockIdx.y * 128;
    const u16* arow = zin + (size_t)(m0 + nl) * 64 + kbA;
    v8f acc[8];
#pragma unroll
    for (int j = 0; j < 8; ++j) acc[j] = acc_zero();
#pragma unroll
    for (int k0 = 0; k0 < 64; k0 += 32) {
        v16bf a = frag_a(arow + k0);
#pragma unroll
        for (int j = 0; j < 8; ++j)
            acc[j] = wmma_bf16(a, frag_b(&wlds[(nbase + j * 16 + nl) * 64 + k0 + kbB]), acc[j]);
    }
    const int mo = (lane >> 4) << 3;
    int nr[8], hwr[8];
#pragma unroll
    for (int r = 0; r < 8; ++r) {
        int pix = m0 + mo + r;
        nr[r] = pix / HW;
        hwr[r] = pix - nr[r] * HW;
    }
#pragma unroll
    for (int j = 0; j < 8; ++j) {
        int o = nbase + j * 16 + nl;
        float s = g[o] * BN_INV, bb = bta[o];
#pragma unroll
        for (int r = 0; r < 8; ++r) {
            size_t idx = ((size_t)nr[r] * 256 + o) * HW + hwr[r];
            out[idx] = fmaxf(acc[j][r] * s + bb + xres[idx], 0.f);
        }
    }
}

// ---------------------------------------------------------------------------
extern "C" void kernel_launch(void* const* d_in, const int* in_sizes, int n_in,
                              void* d_out, int out_size, void* d_ws, size_t ws_size,
                              hipStream_t stream) {
    const float* x     = (const float*)d_in[0];
    const float* w1    = (const float*)d_in[1];
    const float* w3    = (const float*)d_in[2];
    const float* w5    = (const float*)d_in[3];
    const float* w7    = (const float*)d_in[4];
    const float* wc1f  = (const float*)d_in[5];
    const float* g1    = (const float*)d_in[6];
    const float* b1    = (const float*)d_in[7];
    const float* wc21f = (const float*)d_in[8];
    const float* g21   = (const float*)d_in[9];
    const float* b21   = (const float*)d_in[10];
    const float* g22   = (const float*)d_in[11];
    const float* b22   = (const float*)d_in[12];
    const float* wc22f = (const float*)d_in[13];
    const float* g23   = (const float*)d_in[14];
    const float* b23   = (const float*)d_in[15];
    const float* wc2f  = (const float*)d_in[16];
    const float* g2    = (const float*)d_in[17];
    const float* b2    = (const float*)d_in[18];
    const float* wc3f  = (const float*)d_in[19];
    const float* g3    = (const float*)d_in[20];
    const float* b3    = (const float*)d_in[21];

    char* ws = (char*)d_ws;
    size_t off = 0;
    auto take = [&](size_t bytes) -> void* {
        void* p = ws + off;
        off = (off + bytes + 255) & ~(size_t)255;
        return p;
    };
    u16*   tx   = (u16*)take((size_t)NPIX * 256 * 2);   // tconv out, NHWC bf16
    u16*   res2 = (u16*)take((size_t)NPIX * 64 * 2);    // conv1 out
    float* yb   = (float*)take((size_t)NPIX * 16 * 4);  // conv21 out (fp32)
    u16*   corr = (u16*)take((size_t)NPIX * 64 * 2);    // correlation (K padded)
    u16*   out2 = (u16*)take((size_t)NPIX * 64 * 2);
    u16*   zb   = (u16*)take((size_t)NPIX * 64 * 2);
    u16*   wc1  = (u16*)take(64 * 256 * 2);
    u16*   wc21 = (u16*)take(16 * 64 * 2);
    u16*   wc22 = (u16*)take(64 * 64 * 2);
    u16*   wB2  = (u16*)take(9 * 64 * 64 * 2);
    u16*   wc3  = (u16*)take(256 * 64 * 2);

    k_prep<<<144, 256, 0, stream>>>(wc1f, wc21f, wc22f, wc2f, wc3f,
                                    wc1, wc21, wc22, wB2, wc3);
    k_tconv<<<dim3(8, 56, 16), 256, 0, stream>>>(x, w1, w3, w5, w7, tx);
    k_conv1<<<1568, 256, 0, stream>>>(tx, wc1, g1, b1, res2);
    k_conv21<<<1568, 256, 0, stream>>>(res2, wc21, g21, b21, yb);
    k_corr<<<784, 256, 0, stream>>>(yb, g22, b22, corr);
    k_conv22<<<1568, 256, 0, stream>>>(corr, wc22, g23, b23, res2, out2);
    k_conv2<<<1568, 256, 0, stream>>>(res2, wB2, g2, b2, out2, zb);
    k_conv3<<<dim3(1568, 2), 256, 0, stream>>>(zb, wc3, g3, b3, x, (float*)d_out);
}
